// ClueFeat_88218628260002
// MI455X (gfx1250) — compile-verified, matrix operand order checked by reading, not run on hardware
//
#include <hip/hip_runtime.h>
#include <stdint.h>

// ---------------------------------------------------------------------------
// Types
// ---------------------------------------------------------------------------
using h16 = _Float16;
typedef __attribute__((ext_vector_type(16))) _Float16 v16h;
typedef __attribute__((ext_vector_type(8)))  _Float16 v8h;
typedef __attribute__((ext_vector_type(4)))  _Float16 v4hh;
typedef __attribute__((ext_vector_type(8)))  float    v8f;

template <typename X, typename Y> struct same_t { static constexpr bool value = false; };
template <typename X> struct same_t<X, X>      { static constexpr bool value = true;  };

// Problem constants (B=8, T=1024, D=256, K=4 from the reference)
#define NB 8
#define TT 1024
#define DD 256

// ---------------------------------------------------------------------------
// CDNA5 async global->LDS copy (GLOBAL_LOAD_ASYNC_TO_LDS_B128, ASYNCcnt).
// GV addressing: per-lane 64-bit global address, per-lane 32-bit LDS byte
// address in VDST (low 32 bits of a generic pointer into LDS, per ISA §10.2).
// ---------------------------------------------------------------------------
__device__ __forceinline__ void async_ld_b128(void* lds_dst, const void* gsrc) {
  uint32_t lo = (uint32_t)(uintptr_t)lds_dst;
  uint64_t ga = (uint64_t)(uintptr_t)gsrc;
  asm volatile("global_load_async_to_lds_b128 %0, %1, off"
               :: "v"(lo), "v"(ga) : "memory");
}
__device__ __forceinline__ void wait_async0() {
  asm volatile("s_wait_asynccnt 0x0" ::: "memory");
}

// ---------------------------------------------------------------------------
// 0) weight-norm: w = g * v / ||v||  (4 sets of 4)
// ---------------------------------------------------------------------------
__global__ void wnorm_kernel(const float* g0, const float* v0,
                             const float* g1, const float* v1,
                             const float* g2, const float* v2,
                             const float* g3, const float* v3,
                             float* __restrict__ wv) {
  int t = threadIdx.x;
  if (t >= 4) return;
  const float* g; const float* v;
  if (t == 0)      { g = g0; v = v0; }
  else if (t == 1) { g = g1; v = v1; }
  else if (t == 2) { g = g2; v = v2; }
  else             { g = g3; v = v3; }
  float n = sqrtf(v[0]*v[0] + v[1]*v[1] + v[2]*v[2] + v[3]*v[3]);
  float s = g[0] / n;
  #pragma unroll
  for (int i = 0; i < 4; ++i) wv[t * 4 + i] = s * v[i];
}

// ---------------------------------------------------------------------------
// 1) fold clue tensor over its K=4 axis: weighted sum (outw) and/or mean (outm)
//    clue: (B*T, 4, T) fp32 -> out: (B*T, T) f16.  Pure bandwidth kernel.
// ---------------------------------------------------------------------------
__global__ __launch_bounds__(256) void fold_clue_kernel(
    const float* __restrict__ clue, const float* __restrict__ wv,
    h16* __restrict__ outw, h16* __restrict__ outm, long nchunk) {
  long idx = (long)blockIdx.x * blockDim.x + threadIdx.x;
  if (idx >= nchunk) return;
  long bi = idx >> 8;               // T/4 = 256 chunks per (b,i) row
  int  tq = (int)(idx & 255) * 4;
  const float* p = clue + bi * (4L * TT) + tq;
  float4 f0 = *reinterpret_cast<const float4*>(p);
  float4 f1 = *reinterpret_cast<const float4*>(p + TT);
  float4 f2 = *reinterpret_cast<const float4*>(p + 2 * TT);
  float4 f3 = *reinterpret_cast<const float4*>(p + 3 * TT);
  long o = bi * TT + tq;
  if (outw) {
    float w0 = wv[0], w1 = wv[1], w2 = wv[2], w3 = wv[3];
    v4hh h;
    h[0] = (h16)(w0 * f0.x + w1 * f1.x + w2 * f2.x + w3 * f3.x);
    h[1] = (h16)(w0 * f0.y + w1 * f1.y + w2 * f2.y + w3 * f3.y);
    h[2] = (h16)(w0 * f0.z + w1 * f1.z + w2 * f2.z + w3 * f3.z);
    h[3] = (h16)(w0 * f0.w + w1 * f1.w + w2 * f2.w + w3 * f3.w);
    *reinterpret_cast<v4hh*>(outw + o) = h;
  }
  if (outm) {
    v4hh h;
    h[0] = (h16)(0.25f * (f0.x + f1.x + f2.x + f3.x));
    h[1] = (h16)(0.25f * (f0.y + f1.y + f2.y + f3.y));
    h[2] = (h16)(0.25f * (f0.z + f1.z + f2.z + f3.z));
    h[3] = (h16)(0.25f * (f0.w + f1.w + f2.w + f3.w));
    *reinterpret_cast<v4hh*>(outm + o) = h;
  }
}

// ---------------------------------------------------------------------------
// 2) Generic batched WMMA GEMM: C[M,N] = A[M,K] * B[K,N]
//    BT=true  -> B is supplied as N x K row-major (i.e. C = A * B^T)
//    f16 operand tiles staged via GLOBAL_LOAD_ASYNC_TO_LDS_B128 (pure copy);
//    f32 sources converted to f16 while staging; fp32 accumulation.
//    Tile: 64x64 per block (128 threads = 4 waves, 2x2 WMMA tiles each), BK=32.
// ---------------------------------------------------------------------------
template <typename TA, typename TB, bool BT, typename TOUT>
__global__ __launch_bounds__(128) void gemm_wmma_kernel(
    const TA* __restrict__ A, const TB* __restrict__ Bm, TOUT* __restrict__ C,
    int K, long lda, long ldb, long ldc, long strA, long strB, long strC) {
  constexpr int LDP = 48;                       // pitch in halves: 96B, 16B-aligned
  __shared__ h16 As[64 * LDP];
  __shared__ h16 Bs[64 * LDP];                  // Bs[n][k] (column-of-B contiguous in k)

  const int tid  = threadIdx.x;
  const int lane = tid & 31;
  const int wid  = tid >> 5;
  const int wm   = (wid >> 1) * 32;
  const int wn   = (wid & 1) * 32;
  const long m0  = (long)blockIdx.y * 64;
  const long n0  = (long)blockIdx.x * 64;
  const TA* Ab = A  + (long)blockIdx.z * strA + m0 * lda;
  const TB* Bb = Bm + (long)blockIdx.z * strB;
  TOUT*     Cb = C  + (long)blockIdx.z * strC;

  const int lr = lane & 15;
  const int kg = (lane >> 4) * 8;

  v8f acc[2][2] = {};

  for (int k0 = 0; k0 < K; k0 += 32) {
    constexpr bool asyncA = !same_t<TA, float>::value;
    constexpr bool asyncB = BT && !same_t<TB, float>::value;

    // ---- stage A tile: As[r][k] = (h16) A[m0+r][k0+k]
    if constexpr (asyncA) {
      #pragma unroll
      for (int i = 0; i < 2; ++i) {
        int id = tid + i * 128;                  // 256 16-byte chunks
        int r = id >> 2, kq = (id & 3) * 8;
        async_ld_b128(&As[r * LDP + kq], Ab + (long)r * lda + k0 + kq);
      }
    } else {
      if (k0 + 32 < K)                           // prefetch next K-tile
        __builtin_prefetch(Ab + (long)(tid & 63) * lda + k0 + 32, 0, 1);
      #pragma unroll
      for (int i = 0; i < 4; ++i) {
        int id = tid + i * 128;                  // 512 float4 chunks
        int r = id >> 3, kq = (id & 7) * 4;
        float4 f = *reinterpret_cast<const float4*>(Ab + (long)r * lda + k0 + kq);
        h16* d = &As[r * LDP + kq];
        d[0] = (h16)f.x; d[1] = (h16)f.y; d[2] = (h16)f.z; d[3] = (h16)f.w;
      }
    }
    // ---- stage B tile: Bs[n][k] = (h16) Beff[k][n]
    if constexpr (BT) {                          // B given as N x K: row copy like A
      const TB* Bt = Bb + n0 * ldb;
      if constexpr (same_t<TB, float>::value) {
        #pragma unroll
        for (int i = 0; i < 4; ++i) {
          int id = tid + i * 128;
          int r = id >> 3, kq = (id & 7) * 4;
          float4 f = *reinterpret_cast<const float4*>(Bt + (long)r * ldb + k0 + kq);
          h16* d = &Bs[r * LDP + kq];
          d[0] = (h16)f.x; d[1] = (h16)f.y; d[2] = (h16)f.z; d[3] = (h16)f.w;
        }
      } else {
        #pragma unroll
        for (int i = 0; i < 2; ++i) {
          int id = tid + i * 128;
          int r = id >> 2, kq = (id & 3) * 8;
          async_ld_b128(&Bs[r * LDP + kq], Bt + (long)r * ldb + k0 + kq);
        }
      }
    } else {                                     // B given as K x N: transpose on store
      if constexpr (same_t<TB, float>::value) {
        #pragma unroll
        for (int i = 0; i < 4; ++i) {
          int id = tid + i * 128;
          int k = id >> 4, nq = (id & 15) * 4;
          float4 f = *reinterpret_cast<const float4*>(Bb + (long)(k0 + k) * ldb + n0 + nq);
          Bs[(nq + 0) * LDP + k] = (h16)f.x;
          Bs[(nq + 1) * LDP + k] = (h16)f.y;
          Bs[(nq + 2) * LDP + k] = (h16)f.z;
          Bs[(nq + 3) * LDP + k] = (h16)f.w;
        }
      } else {
        #pragma unroll
        for (int i = 0; i < 2; ++i) {
          int id = tid + i * 128;
          int k = id >> 3, nq = (id & 7) * 8;
          v8h h = *reinterpret_cast<const v8h*>(Bb + (long)(k0 + k) * ldb + n0 + nq);
          #pragma unroll
          for (int j = 0; j < 8; ++j) Bs[(nq + j) * LDP + k] = h[j];
        }
      }
    }
    if constexpr (asyncA || asyncB) wait_async0();   // ASYNCcnt drain before barrier
    __syncthreads();

    // ---- fragments per ISA 16-bit A layout: lane half kg, K in [kg,kg+8)∪[16+kg,16+kg+8)
    v16h aF[2], bF[2];
    #pragma unroll
    for (int t = 0; t < 2; ++t) {
      const h16* pa = &As[(wm + t * 16 + lr) * LDP + kg];
      v8h alo = *reinterpret_cast<const v8h*>(pa);
      v8h ahi = *reinterpret_cast<const v8h*>(pa + 16);
      aF[t] = __builtin_shufflevector(alo, ahi, 0,1,2,3,4,5,6,7,8,9,10,11,12,13,14,15);
      const h16* pb = &Bs[(wn + t * 16 + lr) * LDP + kg];
      v8h blo = *reinterpret_cast<const v8h*>(pb);
      v8h bhi = *reinterpret_cast<const v8h*>(pb + 16);
      bF[t] = __builtin_shufflevector(blo, bhi, 0,1,2,3,4,5,6,7,8,9,10,11,12,13,14,15);
    }
    #pragma unroll
    for (int tm = 0; tm < 2; ++tm)
      #pragma unroll
      for (int tn = 0; tn < 2; ++tn)
        acc[tm][tn] = __builtin_amdgcn_wmma_f32_16x16x32_f16(
            false, aF[tm], false, bF[tn], (short)0, acc[tm][tn], false, false);
    __syncthreads();
  }

  // ---- epilogue: C/D layout -> row = (lane>>4)*8 + e, col = lane&15
  const int cr = (lane >> 4) * 8;
  const int cc = lane & 15;
  #pragma unroll
  for (int tm = 0; tm < 2; ++tm)
    #pragma unroll
    for (int tn = 0; tn < 2; ++tn) {
      long gr = m0 + wm + tm * 16 + cr;
      long gc = n0 + wn + tn * 16 + cc;
      #pragma unroll
      for (int e = 0; e < 8; ++e)
        Cb[(gr + e) * ldc + gc] = (TOUT)acc[tm][tn][e];
    }
}

// ---------------------------------------------------------------------------
// 3) sim pair: out = (qwA . kmA^T) * mask_clue(adjA) + (qwB . kmB^T) * mask_clue(adjB)
//    per batch b; qw/km are (T, D) f16; adj/out are (T, T) fp32.
//    Both operand tiles are pure row copies -> async global->LDS staging.
// ---------------------------------------------------------------------------
__global__ __launch_bounds__(128) void simpair_wmma_kernel(
    const h16* __restrict__ qwA, const h16* __restrict__ kmA, const float* __restrict__ adjA,
    const h16* __restrict__ qwB, const h16* __restrict__ kmB, const float* __restrict__ adjB,
    float* __restrict__ out) {
  constexpr int LDP = 48;
  constexpr int KK  = DD;
  __shared__ h16 As[64 * LDP];
  __shared__ h16 Bs[64 * LDP];

  const int tid  = threadIdx.x;
  const int lane = tid & 31;
  const int wid  = tid >> 5;
  const int wm   = (wid >> 1) * 32;
  const int wn   = (wid & 1) * 32;
  const long m0  = (long)blockIdx.y * 64;
  const long n0  = (long)blockIdx.x * 64;
  const long zb  = (long)blockIdx.z;
  const long strQW  = (long)TT * KK;
  const long strADJ = (long)TT * TT;
  const int lr = lane & 15;
  const int kg = (lane >> 4) * 8;

  v8f acc[2][2][2] = {};
  #pragma unroll
  for (int ph = 0; ph < 2; ++ph) {
    const h16* Ab = (ph ? qwB : qwA) + zb * strQW + m0 * KK;
    const h16* Bb = (ph ? kmB : kmA) + zb * strQW + n0 * KK;
    for (int k0 = 0; k0 < KK; k0 += 32) {
      #pragma unroll
      for (int i = 0; i < 2; ++i) {
        int id = tid + i * 128;
        int r = id >> 2, kq = (id & 3) * 8;
        async_ld_b128(&As[r * LDP + kq], Ab + (long)r * KK + k0 + kq);
        async_ld_b128(&Bs[r * LDP + kq], Bb + (long)r * KK + k0 + kq);
      }
      wait_async0();
      __syncthreads();
      v16h aF[2], bF[2];
      #pragma unroll
      for (int t = 0; t < 2; ++t) {
        const h16* pa = &As[(wm + t * 16 + lr) * LDP + kg];
        v8h alo = *reinterpret_cast<const v8h*>(pa);
        v8h ahi = *reinterpret_cast<const v8h*>(pa + 16);
        aF[t] = __builtin_shufflevector(alo, ahi, 0,1,2,3,4,5,6,7,8,9,10,11,12,13,14,15);
        const h16* pb = &Bs[(wn + t * 16 + lr) * LDP + kg];
        v8h blo = *reinterpret_cast<const v8h*>(pb);
        v8h bhi = *reinterpret_cast<const v8h*>(pb + 16);
        bF[t] = __builtin_shufflevector(blo, bhi, 0,1,2,3,4,5,6,7,8,9,10,11,12,13,14,15);
      }
      #pragma unroll
      for (int tm = 0; tm < 2; ++tm)
        #pragma unroll
        for (int tn = 0; tn < 2; ++tn)
          acc[ph][tm][tn] = __builtin_amdgcn_wmma_f32_16x16x32_f16(
              false, aF[tm], false, bF[tn], (short)0, acc[ph][tm][tn], false, false);
      __syncthreads();
    }
  }
  const float* aA = adjA + zb * strADJ;
  const float* aB = adjB + zb * strADJ;
  float*       ob = out  + zb * strADJ;
  const int cr = (lane >> 4) * 8;
  const int cc = lane & 15;
  #pragma unroll
  for (int tm = 0; tm < 2; ++tm)
    #pragma unroll
    for (int tn = 0; tn < 2; ++tn) {
      long gr = m0 + wm + tm * 16 + cr;
      long gc = n0 + wn + tn * 16 + cc;
      #pragma unroll
      for (int e = 0; e < 8; ++e) {
        long idx = (gr + e) * TT + gc;
        float vA = aA[idx], vB = aB[idx];
        float mA = (vA == 0.f) ? 0.f : (vA > 0.f ? 1.f : vA);
        float mB = (vB == 0.f) ? 0.f : (vB > 0.f ? 1.f : vB);
        ob[idx] = acc[0][tm][tn][e] * mA + acc[1][tm][tn][e] * mB;
      }
    }
}

// ---------------------------------------------------------------------------
// 4) L2-normalize rows of 256 fp32 -> f16
// ---------------------------------------------------------------------------
__global__ __launch_bounds__(64) void l2norm_kernel(const float* __restrict__ in,
                                                    h16* __restrict__ out) {
  long row = blockIdx.x;
  int tid  = threadIdx.x;
  const float* p = in + row * DD + tid * 4;
  float4 f = *reinterpret_cast<const float4*>(p);
  __shared__ float red[64];
  red[tid] = f.x * f.x + f.y * f.y + f.z * f.z + f.w * f.w;
  __syncthreads();
  #pragma unroll
  for (int s = 32; s > 0; s >>= 1) {
    if (tid < s) red[tid] += red[tid + s];
    __syncthreads();
  }
  float inv = 1.f / fmaxf(sqrtf(red[0]), 1e-12f);
  v4hh o;
  o[0] = (h16)(f.x * inv); o[1] = (h16)(f.y * inv);
  o[2] = (h16)(f.z * inv); o[3] = (h16)(f.w * inv);
  *reinterpret_cast<v4hh*>(out + row * DD + tid * 4) = o;
}

// ---------------------------------------------------------------------------
// 5) fused mask + cross-sigmoid-mask + dual softmax (one block per (b,i) row)
// ---------------------------------------------------------------------------
__device__ inline float r2_chain(float m) {
  float m1 = ((m >= -0.1f) && (m <= 0.1f) && (m != 0.f)) ? 1.f : m;
  float m2 = ((m1 >= 0.1f) && (m1 != 1.f)) ? 0.f : m1;
  return ((m2 < -0.1f) && (m2 != 1.f)) ? 0.f : m2;
}
__device__ inline float mask_att_f(float m) {
  return (m == 0.f) ? -1e22f : (m > 0.f ? 0.f : m);
}

__global__ __launch_bounds__(256) void mask_softmax_kernel(
    const float* __restrict__ sA, const float* __restrict__ sB,
    const float* __restrict__ adjA, const float* __restrict__ adjB,
    h16* __restrict__ att, h16* __restrict__ att_t) {
  const long row  = blockIdx.x;                 // b*T + i
  const int  tid  = threadIdx.x;
  const long base = row * TT + tid * 4;
  float4 a4 = *reinterpret_cast<const float4*>(sA   + base);
  float4 b4 = *reinterpret_cast<const float4*>(sB   + base);
  float4 mA = *reinterpret_cast<const float4*>(adjA + base);
  float4 mB = *reinterpret_cast<const float4*>(adjB + base);
  float av[4] = {a4.x, a4.y, a4.z, a4.w};
  float bv[4] = {b4.x, b4.y, b4.z, b4.w};
  float mAv[4] = {mA.x, mA.y, mA.z, mA.w};
  float mBv[4] = {mB.x, mB.y, mB.z, mB.w};
  float za[4], zb[4];
  #pragma unroll
  for (int j = 0; j < 4; ++j) {
    float a = av[j], b = bv[j];
    float r1a = (fabsf(a) > 0.9f) ? 1.f : 0.f;
    float r1b = (fabsf(b) > 0.9f) ? 1.f : 0.f;
    float sga = 1.f / (1.f + expf(-a));
    float sgb = 1.f / (1.f + expf(-b));
    float sim_mask   = r1a * r2_chain(b) * sga; // -> att_t logits
    float sim_t_mask = r1b * r2_chain(a) * sgb; // -> att logits
    za[j] = a + sim_t_mask + mask_att_f(mAv[j]);
    zb[j] = b + sim_mask   + mask_att_f(mBv[j]);
  }
  __shared__ float redA[256];
  __shared__ float redB[256];
  redA[tid] = fmaxf(fmaxf(za[0], za[1]), fmaxf(za[2], za[3]));
  redB[tid] = fmaxf(fmaxf(zb[0], zb[1]), fmaxf(zb[2], zb[3]));
  __syncthreads();
  #pragma unroll
  for (int s = 128; s > 0; s >>= 1) {
    if (tid < s) {
      redA[tid] = fmaxf(redA[tid], redA[tid + s]);
      redB[tid] = fmaxf(redB[tid], redB[tid + s]);
    }
    __syncthreads();
  }
  float mxA = redA[0], mxB = redB[0];
  __syncthreads();
  float ea[4], eb[4], sa = 0.f, sb = 0.f;
  #pragma unroll
  for (int j = 0; j < 4; ++j) {
    ea[j] = expf(za[j] - mxA); sa += ea[j];
    eb[j] = expf(zb[j] - mxB); sb += eb[j];
  }
  redA[tid] = sa; redB[tid] = sb;
  __syncthreads();
  #pragma unroll
  for (int s = 128; s > 0; s >>= 1) {
    if (tid < s) { redA[tid] += redA[tid + s]; redB[tid] += redB[tid + s]; }
    __syncthreads();
  }
  float ia = 1.f / redA[0], ib = 1.f / redB[0];
  v4hh oa, obh;
  #pragma unroll
  for (int j = 0; j < 4; ++j) { oa[j] = (h16)(ea[j] * ia); obh[j] = (h16)(eb[j] * ib); }
  *reinterpret_cast<v4hh*>(att   + base) = oa;
  *reinterpret_cast<v4hh*>(att_t + base) = obh;
}

// ---------------------------------------------------------------------------
// Host launcher
// ---------------------------------------------------------------------------
extern "C" void kernel_launch(void* const* d_in, const int* in_sizes, int n_in,
                              void* d_out, int out_size, void* d_ws, size_t ws_size,
                              hipStream_t stream) {
  (void)in_sizes; (void)n_in; (void)out_size; (void)ws_size;

  const long BT  = (long)NB * TT;         // 8192
  const long BTT = BT * TT;               // 8,388,608
  const long BTD = BT * DD;               // 2,097,152

  const float* x         = (const float*)d_in[0];
  const float* x_t       = (const float*)d_in[1];
  const float* adj_sc    = (const float*)d_in[2];
  const float* adj_ec    = (const float*)d_in[3];
  const float* adj_sc_t  = (const float*)d_in[4];
  const float* adj_sec_t = (const float*)d_in[5];
  const float* sclue_1   = (const float*)d_in[6];
  const float* eclue_1   = (const float*)d_in[7];
  const float* sclue_1_t = (const float*)d_in[8];
  const float* eclue_1_t = (const float*)d_in[9];
  const float* w_q  = (const float*)d_in[10];
  const float* w_k  = (const float*)d_in[11];
  const float* w_v  = (const float*)d_in[12];
  const float* w_qt = (const float*)d_in[13];
  const float* w_kt = (const float*)d_in[14];
  const float* ss_g  = (const float*)d_in[15];
  const float* ss_v  = (const float*)d_in[16];
  const float* ii_g  = (const float*)d_in[17];
  const float* ii_v  = (const float*)d_in[18];
  const float* sst_g = (const float*)d_in[19];
  const float* sst_v = (const float*)d_in[20];
  const float* sit_g = (const float*)d_in[21];
  const float* sit_v = (const float*)d_in[22];
  float* out = (float*)d_out;

  // ---- workspace carving (deterministic) ----
  uint8_t* w8 = (uint8_t*)d_ws;
  size_t off = 0;
  auto carve = [&](size_t bytes) -> void* {
    void* p = w8 + off;
    off += (bytes + 255) & ~(size_t)255;
    return p;
  };
  float* wvec = (float*)carve(16 * sizeof(float));
  h16* cq1 = (h16*)carve(BTT * 2); h16* cq2 = (h16*)carve(BTT * 2);
  h16* cq3 = (h16*)carve(BTT * 2); h16* cq4 = (h16*)carve(BTT * 2);
  h16* ck1 = (h16*)carve(BTT * 2); h16* ck2 = (h16*)carve(BTT * 2);
  h16* ck3 = (h16*)carve(BTT * 2); h16* ck4 = (h16*)carve(BTT * 2);
  h16* qbuf  = (h16*)carve(BTD * 2);
  h16* kbuf  = (h16*)carve(BTD * 2);
  h16* vbuf  = (h16*)carve(BTD * 2);
  h16* qtbuf = (h16*)carve(BTD * 2);
  h16* ktbuf = (h16*)carve(BTD * 2);
  float* proj32 = (float*)carve(BTD * 4);
  h16* qw1 = (h16*)carve(BTD * 2); h16* km1 = (h16*)carve(BTD * 2);
  h16* qw2 = (h16*)carve(BTD * 2); h16* km2 = (h16*)carve(BTD * 2);
  h16* qw3 = (h16*)carve(BTD * 2); h16* km3 = (h16*)carve(BTD * 2);
  h16* qw4 = (h16*)carve(BTD * 2); h16* km4 = (h16*)carve(BTD * 2);
  float* sAbuf = (float*)carve(BTT * 4);
  float* sBbuf = (float*)carve(BTT * 4);
  h16* attb  = (h16*)carve(BTT * 2);
  h16* attbt = (h16*)carve(BTT * 2);

  // ---- 0) weight norm ----
  wnorm_kernel<<<1, 4, 0, stream>>>(ss_g, ss_v, ii_g, ii_v, sst_g, sst_v, sit_g, sit_v, wvec);

  // ---- 1) clue folds (each streams one 134MB tensor once) ----
  long nchunk = BTT / 4;
  dim3 fb((unsigned)((nchunk + 255) / 256));
  fold_clue_kernel<<<fb, 256, 0, stream>>>(sclue_1,   wvec + 0,  cq1, ck1, nchunk);
  fold_clue_kernel<<<fb, 256, 0, stream>>>(eclue_1,   wvec + 4,  cq2, ck2, nchunk);
  fold_clue_kernel<<<fb, 256, 0, stream>>>(sclue_1_t, wvec + 8,  cq3, ck3, nchunk);
  fold_clue_kernel<<<fb, 256, 0, stream>>>(sclue_1_t, wvec + 12, cq4, (h16*)nullptr, nchunk);
  fold_clue_kernel<<<fb, 256, 0, stream>>>(eclue_1_t, (const float*)nullptr, (h16*)nullptr, ck4, nchunk);

  // ---- 2) projections: (B*T,256) @ (256,256); l2-norm for q/k/q_t/k_t ----
  dim3 g1(DD / 64, (unsigned)(BT / 64), 1);
  dim3 nrm((unsigned)BT);
  gemm_wmma_kernel<float, float, false, float><<<g1, 128, 0, stream>>>(
      x, w_q, proj32, DD, DD, DD, DD, 0, 0, 0);
  l2norm_kernel<<<nrm, 64, 0, stream>>>(proj32, qbuf);
  gemm_wmma_kernel<float, float, false, float><<<g1, 128, 0, stream>>>(
      x, w_k, proj32, DD, DD, DD, DD, 0, 0, 0);
  l2norm_kernel<<<nrm, 64, 0, stream>>>(proj32, kbuf);
  gemm_wmma_kernel<float, float, false, float><<<g1, 128, 0, stream>>>(
      x_t, w_qt, proj32, DD, DD, DD, DD, 0, 0, 0);
  l2norm_kernel<<<nrm, 64, 0, stream>>>(proj32, qtbuf);
  gemm_wmma_kernel<float, float, false, float><<<g1, 128, 0, stream>>>(
      x_t, w_kt, proj32, DD, DD, DD, DD, 0, 0, 0);
  l2norm_kernel<<<nrm, 64, 0, stream>>>(proj32, ktbuf);
  gemm_wmma_kernel<float, float, false, h16><<<g1, 128, 0, stream>>>(
      x, w_v, vbuf, DD, DD, DD, DD, 0, 0, 0);

  // ---- 3) qw/km GEMMs: (T,T) f16 @ (T,D) f16, batched over B ----
  dim3 g2(DD / 64, TT / 64, NB);
  const long sTT = (long)TT * TT, sTD = (long)TT * DD;
  gemm_wmma_kernel<h16, h16, false, h16><<<g2, 128, 0, stream>>>(cq1, qbuf,  qw1, TT, TT, DD, DD, sTT, sTD, sTD);
  gemm_wmma_kernel<h16, h16, false, h16><<<g2, 128, 0, stream>>>(ck1, qbuf,  km1, TT, TT, DD, DD, sTT, sTD, sTD);
  gemm_wmma_kernel<h16, h16, false, h16><<<g2, 128, 0, stream>>>(cq2, kbuf,  qw2, TT, TT, DD, DD, sTT, sTD, sTD);
  gemm_wmma_kernel<h16, h16, false, h16><<<g2, 128, 0, stream>>>(ck2, kbuf,  km2, TT, TT, DD, DD, sTT, sTD, sTD);
  gemm_wmma_kernel<h16, h16, false, h16><<<g2, 128, 0, stream>>>(cq3, qtbuf, qw3, TT, TT, DD, DD, sTT, sTD, sTD);
  gemm_wmma_kernel<h16, h16, false, h16><<<g2, 128, 0, stream>>>(ck3, qtbuf, km3, TT, TT, DD, DD, sTT, sTD, sTD);
  gemm_wmma_kernel<h16, h16, false, h16><<<g2, 128, 0, stream>>>(cq4, ktbuf, qw4, TT, TT, DD, DD, sTT, sTD, sTD);
  gemm_wmma_kernel<h16, h16, false, h16><<<g2, 128, 0, stream>>>(ck4, ktbuf, km4, TT, TT, DD, DD, sTT, sTD, sTD);

  // ---- 4) sA / sB with mask_clue epilogue ----
  dim3 g3(TT / 64, TT / 64, NB);
  simpair_wmma_kernel<<<g3, 128, 0, stream>>>(qw1, km1, adj_sc,   qw2, km2, adj_ec,    sAbuf);
  simpair_wmma_kernel<<<g3, 128, 0, stream>>>(qw3, km3, adj_sc_t, qw4, km4, adj_sec_t, sBbuf);

  // ---- 5) fused masks + dual softmax ----
  mask_softmax_kernel<<<(unsigned)BT, 256, 0, stream>>>(sAbuf, sBbuf, adj_sc, adj_sec_t, attb, attbt);

  // ---- 6) att @ v -> d_out (out, then out_t; both use v per the reference) ----
  dim3 g4(DD / 64, TT / 64, NB);
  gemm_wmma_kernel<h16, h16, false, float><<<g4, 128, 0, stream>>>(
      attb,  vbuf, out,        TT, TT, DD, DD, sTT, sTD, sTD);
  gemm_wmma_kernel<h16, h16, false, float><<<g4, 128, 0, stream>>>(
      attbt, vbuf, out + BTD,  TT, TT, DD, DD, sTT, sTD, sTD);
}
